// TSDCD_75866302317177
// MI455X (gfx1250) — matrix operand order repaired and checked by reading, not compiled, not run on hardware
//
#include <hip/hip_runtime.h>
#include <hip/hip_bf16.h>
#include <stdint.h>

typedef __attribute__((ext_vector_type(16))) _Float16 v16h;
typedef __attribute__((ext_vector_type(8)))  float    v8f;

#define D_TOT 320
#define DX    256
#define DZ    64
#define HH    16
#define BB    1024
#define KT    16          // k-variables per workgroup
#define BT    16          // batch rows per workgroup
#define JTI   32          // j (K-dim) tile
#define NJT   (D_TOT/JTI) // 10
#define AV_STRIDE 324     // padded: stride-324 -> 16 distinct banks
#define U_STRIDE  520     // 512 + TDM pad (4 DWORDs per 256) -> 8-bank spread
#define NEG_SLOPE 0.01f
#define HALF_LOG2PI 0.9189385332046727f
#define NPART (64*20)

#if defined(__AMDGCN__) && __has_builtin(__builtin_amdgcn_tensor_load_to_lds)
#define USE_TDM 1
#else
#define USE_TDM 0
#endif

__device__ __forceinline__ unsigned lds_off(const void* p) {
  return (unsigned)(uintptr_t)p;   // low 32 bits of generic ptr = LDS byte offset
}

#if USE_TDM
typedef unsigned int u32x4 __attribute__((ext_vector_type(4)));
typedef int          i32x4 __attribute__((ext_vector_type(4)));
typedef int          i32x8 __attribute__((ext_vector_type(8)));

// TDM D# (ISA 8.3-8.6): 3D tile; dim0 = 16 k-floats (contig), dim1 = 32 j rows
// (stride 320), dim2 = 16 batch rows (stride 102400). LDS pad: 4 DWORDs per
// 256 DWORDs -> per-batch-row stride 520 DWORDs, +4 shift for j>=16.
__device__ __forceinline__ void tdm_load_u(const float* u, int bBase, int kBase,
                                           int jt, unsigned ldsByteOff) {
  unsigned long long ga = (unsigned long long)(uintptr_t)u +
      4ull * ((unsigned long long)bBase * (D_TOT * D_TOT) +
              (unsigned long long)jt * JTI * D_TOT + (unsigned long long)kBase);
  u32x4 g0 = { 1u,                                   // count=1 (valid descriptor)
               ldsByteOff,                           // lds_addr
               (unsigned)(ga & 0xFFFFFFFFu),         // global_addr[31:0]
               (unsigned)((ga >> 32) & 0x1FFFFFFu) | 0x80000000u }; // [56:32] | type=2
  // g1[0]: data_size=2(4B), pad_enable, pad_interval=7(256 DW), pad_amount=3(4 DW)
  i32x8 g1 = { (int)((2u << 16) | (1u << 20) | (7u << 22) | (3u << 25)),
               (int)(16u << 16),                     // tensor_dim0 = 16
               (int)(32u << 16),                     // tensor_dim1 = 32
               (int)(16u << 16),                     // tile_dim0 = 16
               (int)(32u | (16u << 16)),             // tile_dim1=32, tile_dim2=16
               (int)D_TOT,                           // tensor_dim0_stride = 320
               (int)0x90000000u,                     // tensor_dim1_stride[15:0]=0x9000
               1 };                                  // tensor_dim1_stride[47:16]=1 (102400)
  i32x4 g2 = { 16, 0, 0, 0 };                        // tensor_dim2 = 16
  i32x4 g3 = { 0, 0, 0, 0 };
#if __clang_major__ >= 23
  i32x8 g4 = { 0, 0, 0, 0, 0, 0, 0, 0 };
  __builtin_amdgcn_tensor_load_to_lds(g0, g1, g2, g3, g4, 0);
#else
  __builtin_amdgcn_tensor_load_to_lds(g0, g1, g2, g3, 0);
#endif
}

__device__ __forceinline__ void wait_tensor_le1() {
#if __has_builtin(__builtin_amdgcn_s_wait_tensorcnt)
  __builtin_amdgcn_s_wait_tensorcnt(1);
#else
  asm volatile("s_wait_tensorcnt 0x1" ::: "memory");
#endif
}
__device__ __forceinline__ void wait_tensor_0() {
#if __has_builtin(__builtin_amdgcn_s_wait_tensorcnt)
  __builtin_amdgcn_s_wait_tensorcnt(0);
#else
  asm volatile("s_wait_tensorcnt 0x0" ::: "memory");
#endif
}
#endif // USE_TDM

// ---- A fragment (16x32 f16, ISA 7.12.2): lane<16: M=lane,Kbase=0; else Kbase=8.
// VGPR v halves i=2v+hb: K = (v<4 ? 2v : 16+2(v-4)) + hb + Kbase
__device__ __forceinline__ v16h build_a_u(const float* sUc, const float* sE,
                                          const float* sAV, int kk, int jt, int lane) {
  int m = lane & 15;
  int base = (lane >> 4) << 3;
  union { v16h v; _Float16 h[16]; } a;
#pragma unroll
  for (int i = 0; i < 16; ++i) {
    int vv = i >> 1, hb = i & 1;
    int jl = ((vv < 4) ? (2 * vv) : (16 + 2 * (vv - 4))) + hb + base;
    float uu = sUc[m * U_STRIDE + (jl << 4) + kk + ((jl >> 4) << 2)];
    float E  = sE[(jl << 4) + kk];
    float av = sAV[m * AV_STRIDE + jt * JTI + jl];
    // sigmoid(mask_param + logit(u)) = u*e^m / (u*e^m + 1 - u); E carries fixed_mask
    float t = uu * E;
    float adj = t * __builtin_amdgcn_rcpf(t + 1.0f - uu);
    a.h[i] = (_Float16)(adj * av);
  }
  return a.v;
}

// ---------------- kernel 1: E = fixed_mask * exp(mask_param) ----------------

__global__ __launch_bounds__(256) void k_E(const float* __restrict__ mp,
                                           float* __restrict__ E) {
  int idx = blockIdx.x * 256 + threadIdx.x;      // D*D = 102400
  int j = idx / D_TOT, k = idx - j * D_TOT;
  float fm = (k < DZ) ? ((j == k) ? 1.f : 0.f) : ((j == k) ? 0.f : 1.f);
  E[idx] = (fm != 0.f) ? __expf(mp[idx]) : 0.f;
}

// ---------------- kernel 2: fused adjacency + 3-layer MLP + logp ------------

__global__ __launch_bounds__(256) void k_main(
    const float* __restrict__ x, const float* __restrict__ z,
    const float* __restrict__ u, const float* __restrict__ W0,
    const float* __restrict__ b0g, const float* __restrict__ W1,
    const float* __restrict__ b1g, const float* __restrict__ W2,
    const float* __restrict__ b2g, const float* __restrict__ logvar,
    const float* __restrict__ Ews, float* __restrict__ partials,
    float* __restrict__ out) {
  __shared__ float sAV[BT * AV_STRIDE];
  __shared__ float sU[2][BT * U_STRIDE];                 // TDM double buffer
  __shared__ float sE[JTI * KT];
  __shared__ alignas(32) _Float16 sW0f[KT * 32 * 16];    // B-frag layout [k][lane][i]
  __shared__ alignas(32) _Float16 sW1f[KT * 32 * 16];    // B-frag layout (K>=16 zero)
  __shared__ alignas(32) _Float16 sH1f[8 * 32 * 16];     // A-frag layout per wave
  __shared__ float sB0[KT * HH];
  __shared__ float sB1[KT * HH];
  __shared__ float sW2l[KT * HH];
  __shared__ float sB2[KT];
  __shared__ float sRed[8];

  const int tid   = threadIdx.x;
  const int lane  = tid & 31;
  const int w     = tid >> 5;
  const int bBase = blockIdx.x * BT;
  const int kBase = blockIdx.y * KT;

#if USE_TDM
  if (tid < 32) tdm_load_u(u, bBase, kBase, 0, lds_off(&sU[0][0]));  // prologue DMA
#endif

  // ---- one-time staging ----
#pragma unroll
  for (int it = 0; it < 20; ++it) {                      // all_var 16x320
    int idx = tid + it * 256;
    int bb = idx / D_TOT, j = idx - bb * D_TOT;
    int gb = bBase + bb;
    float v = (j < DZ) ? z[gb * DZ + j] : x[gb * DX + (j - DZ)];
    sAV[bb * AV_STRIDE + j] = v;
  }
#pragma unroll
  for (int it = 0; it < 16; ++it) sH1f[tid + it * 256] = (_Float16)0.0f;
#pragma unroll
  for (int it = 0; it < 32; ++it) {                      // W1 -> B-frag layout + zero pad
    int idx = tid + it * 256;
    int hc = idx & 15; int rest = idx >> 4; int hr = rest & 31; int kk = rest >> 5;
    _Float16 v = (_Float16)0.0f;
    if (hr < 16) v = (_Float16)W1[((kBase + kk) * 16 + hr) * 16 + hc];
    int lane_d = hc + (((hr >> 3) & 1) << 4);
    sW1f[(kk * 32 + lane_d) * 16 + ((hr & 7) + ((hr >> 4) << 3))] = v;
  }
  if (tid < KT * HH) {
    int kk = tid >> 4, h = tid & 15;
    sB0[tid]  = b0g[(kBase + kk) * 16 + h];
    sB1[tid]  = b1g[(kBase + kk) * 16 + h];
    sW2l[tid] = W2[(kBase + kk) * 16 + h];
  }
  if (tid < KT) sB2[tid] = b2g[kBase + tid];

  v8f acc0 = {}; v8f acc1 = {};

  // ---- main loop over j tiles (K dim) ----
  for (int jt = 0; jt < NJT; ++jt) {
    const int cb = jt & 1;
    __syncthreads();   // prior compute done: safe to overwrite sE/sW0f and sU[cb^1]

#if USE_TDM
    if (tid < 32) {
      if (jt + 1 < NJT) {
        tdm_load_u(u, bBase, kBase, jt + 1, lds_off(&sU[cb ^ 1][0]));
        wait_tensor_le1();                 // tile jt complete, jt+1 in flight
      } else {
        wait_tensor_0();
      }
    }
#else
#pragma unroll
    for (int it = 0; it < 8; ++it) {       // fallback: coalesced float4 staging
      int idx = tid + it * 256;
      int kq = idx & 3; int rest = idx >> 2; int jj = rest & 31; int bb = rest >> 5;
      float4 v = *((const float4*)(u + ((size_t)(bBase + bb) * D_TOT + (jt * JTI + jj)) * D_TOT + kBase) + kq);
      *(float4*)(&sU[cb][bb * U_STRIDE + jj * 16 + kq * 4 + ((jj >= 16) ? 4 : 0)]) = v;
    }
#endif
#pragma unroll
    for (int it = 0; it < 2; ++it) {       // E tile 32x16
      int idx = tid + it * 256;
      int jj = idx >> 4; int kk = idx & 15;
      sE[jj * KT + kk] = Ews[(jt * JTI + jj) * D_TOT + kBase + kk];
    }
#pragma unroll
    for (int it = 0; it < 8; ++it) {       // W0 tile f32 -> f16, B-frag layout
      int idx = tid + it * 256;
      int hq = idx & 3; int rest = idx >> 2; int jj = rest & 31; int kk = rest >> 5;
      float4 v = *((const float4*)(W0 + ((size_t)(kBase + kk) * D_TOT + jt * JTI + jj) * HH) + hq);
      int i = (jj & 7) | ((jj >> 4) << 3), hi16 = ((jj >> 3) & 1) << 4;
      _Float16* dst = &sW0f[kk * 512 + i];
      dst[(hq * 4 + 0 + hi16) * 16] = (_Float16)v.x;
      dst[(hq * 4 + 1 + hi16) * 16] = (_Float16)v.y;
      dst[(hq * 4 + 2 + hi16) * 16] = (_Float16)v.z;
      dst[(hq * 4 + 3 + hi16) * 16] = (_Float16)v.w;
    }
    __syncthreads();

    const float* sUc = sU[cb];
    {
      int kk = 2 * w;
      v16h a = build_a_u(sUc, sE, sAV, kk, jt, lane);
      v16h b = *(const v16h*)(sW0f + kk * 512 + lane * 16);   // 2x ds_load_b128
      acc0 = __builtin_amdgcn_wmma_f32_16x16x32_f16(false, a, false, b, (short)0, acc0, false, false);
      kk = 2 * w + 1;
      a = build_a_u(sUc, sE, sAV, kk, jt, lane);
      b = *(const v16h*)(sW0f + kk * 512 + lane * 16);
      acc1 = __builtin_amdgcn_wmma_f32_16x16x32_f16(false, a, false, b, (short)0, acc1, false, false);
    }
  }

  // ---- tail: bias+leaky, layer2 WMMA (zero-padded K), layer3 dot, logp ----
  const int n  = lane & 15;
  const int hi = lane >> 4;
  float lacc = 0.f;
#pragma unroll
  for (int t = 0; t < 2; ++t) {
    int kk = 2 * w + t;
    v8f acc = t ? acc1 : acc0;
    float bias0 = sB0[kk * 16 + n];
#pragma unroll
    for (int r = 0; r < 8; ++r) {
      float v = acc[r] + bias0;
      v = (v > 0.f) ? v : NEG_SLOPE * v;
      // h1 col n of row M -> A-frag position [lane_d = M + 16*(n>=8)][i = n&7]
      int M = r + (hi << 3);
      sH1f[w * 512 + (M + ((n >= 8) << 4)) * 16 + (n & 7)] = (_Float16)v;
    }
    v16h a2 = *(const v16h*)(sH1f + w * 512 + lane * 16);     // same-wave RAW, in-order LDS
    v16h b2 = *(const v16h*)(sW1f + kk * 512 + lane * 16);
    v8f c2 = {};
    c2 = __builtin_amdgcn_wmma_f32_16x16x32_f16(false, a2, false, b2, (short)0, c2, false, false);

    float bias1 = sB1[kk * 16 + n];
    float w2v   = sW2l[kk * 16 + n];
    int   kG    = kBase + kk;
    float rstd  = __expf(-0.5f * logvar[kG]);
#pragma unroll
    for (int r = 0; r < 8; ++r) {
      float v = c2[r] + bias1;
      v = (v > 0.f) ? v : NEG_SLOPE * v;
      float p = v * w2v;                   // dot over h within 16-lane halves
      p += __shfl_xor(p, 1, 32);
      p += __shfl_xor(p, 2, 32);
      p += __shfl_xor(p, 4, 32);
      p += __shfl_xor(p, 8, 32);
      if (n == 0) {
        int M = r + (hi << 3);
        float mu = p + sB2[kk];
        out[1 + (size_t)(bBase + M) * D_TOT + kG] = mu;
        float av = sAV[M * AV_STRIDE + kG];
        float diff = (av - mu) * rstd;
        lacc += -0.5f * diff * diff;
      }
    }
  }
#pragma unroll
  for (int m = 1; m < 32; m <<= 1) lacc += __shfl_xor(lacc, m, 32);
  if (lane == 0) sRed[w] = lacc;
  __syncthreads();
  if (tid == 0) {
    float s = 0.f;
#pragma unroll
    for (int i = 0; i < 8; ++i) s += sRed[i];
    partials[blockIdx.y * gridDim.x + blockIdx.x] = s;
  }
}

// ---------------- kernel 3: std + deterministic recons reduction ------------

__global__ __launch_bounds__(320) void k_final(const float* __restrict__ logvar,
                                               const float* __restrict__ partials,
                                               float* __restrict__ out) {
  __shared__ float cs[D_TOT];
  int t = threadIdx.x;
  float s = __expf(0.5f * logvar[t]);
  out[1 + (size_t)BB * D_TOT + t] = s;            // std
  cs[t] = -__logf(s) - HALF_LOG2PI;
  __syncthreads();
  if (t == 0) {
    float c = 0.f;
    for (int i = 0; i < D_TOT; ++i) c += cs[i];
    float p = 0.f;
    for (int i = 0; i < NPART; ++i) p += partials[i];
    out[0] = c + p * (1.0f / (float)BB);          // recons
  }
}

// ---------------------------------------------------------------------------

extern "C" void kernel_launch(void* const* d_in, const int* in_sizes, int n_in,
                              void* d_out, int out_size, void* d_ws, size_t ws_size,
                              hipStream_t stream) {
  const float* x      = (const float*)d_in[0];
  const float* z      = (const float*)d_in[1];
  const float* u      = (const float*)d_in[2];
  const float* mp     = (const float*)d_in[3];
  const float* W0     = (const float*)d_in[4];
  const float* b0     = (const float*)d_in[5];
  const float* W1     = (const float*)d_in[6];
  const float* b1     = (const float*)d_in[7];
  const float* W2     = (const float*)d_in[8];
  const float* b2     = (const float*)d_in[9];
  const float* logvar = (const float*)d_in[10];
  float* out = (float*)d_out;

  float* Ews      = (float*)d_ws;             // D*D floats (409,600 B)
  float* partials = Ews + D_TOT * D_TOT;      // 1280 floats

  k_E<<<dim3((D_TOT * D_TOT) / 256), dim3(256), 0, stream>>>(mp, Ews);
  k_main<<<dim3(BB / BT, D_TOT / KT), dim3(256), 0, stream>>>(
      x, z, u, W0, b0, W1, b1, W2, b2, logvar, Ews, partials, out);
  k_final<<<dim3(1), dim3(D_TOT), 0, stream>>>(logvar, partials, out);
}
